// SyntaxGAT2_19653770347183
// MI455X (gfx1250) — compile-verified, matrix operand order checked by reading, not run on hardware
//
#include <hip/hip_runtime.h>
#include <hip/hip_bf16.h>

typedef __attribute__((ext_vector_type(16))) __bf16 v16bf;
typedef __attribute__((ext_vector_type(8)))  __bf16 v8bf;
typedef __attribute__((ext_vector_type(8)))  float  v8f;

// ---------- bf16 helpers (storage type: unsigned short) ----------
__device__ __forceinline__ unsigned short f2bf(float f) {
  unsigned u = __float_as_uint(f);
  unsigned r = (u + 0x7FFFu + ((u >> 16) & 1u)) >> 16;  // round-to-nearest-even
  return (unsigned short)r;
}
__device__ __forceinline__ float bf2f(unsigned short h) {
  return __uint_as_float(((unsigned)h) << 16);
}
__device__ __forceinline__ float4 ld4f(const float* p) { return *(const float4*)p; }
__device__ __forceinline__ float4 ld4bf(const unsigned short* p) {
  unsigned long long u = *(const unsigned long long*)p;
  float4 r;
  r.x = bf2f((unsigned short)(u));
  r.y = bf2f((unsigned short)(u >> 16));
  r.z = bf2f((unsigned short)(u >> 32));
  r.w = bf2f((unsigned short)(u >> 48));
  return r;
}
__device__ __forceinline__ void fma4(float4& a, float s, const float4& b) {
  a.x = fmaf(s, b.x, a.x); a.y = fmaf(s, b.y, a.y);
  a.z = fmaf(s, b.z, a.z); a.w = fmaf(s, b.w, a.w);
}

// ---------- weight repack: row-major f32 [K,N] -> WMMA B-fragment bf16 ----------
// Layout: Wp[((kt*2 + half)*N + n)*16 + e]
//   half=0 (lanes 0..15):  e=0..7 -> k=kt*32+e        ; e=8..15 -> k=kt*32+8+e  (16..23)
//   half=1 (lanes 16..31): e=0..7 -> k=kt*32+8+e (8..15); e=8..15 -> k=kt*32+e+16 (24..31)
__global__ void pack_w_kernel(const float* __restrict__ W,
                              unsigned short* __restrict__ Wp,
                              int K, int N, int total) {
  int idx = blockIdx.x * blockDim.x + threadIdx.x;
  if (idx >= total) return;
  int e = idx & 15;
  int t = idx >> 4;
  int n = t % N; t /= N;
  int half = t & 1;
  int kt = t >> 1;
  int k = (half == 0) ? ((e < 8) ? e : (8 + e))
                      : ((e < 8) ? (8 + e) : (e + 16));
  k += kt * 32;
  Wp[idx] = f2bf(W[(long long)k * N + n]);
}

// ---------- fused gather: X = gether(H) + H = 2H + adj*diag + t*(k±1) + a*(l±1) ----------
template <bool IN_BF16>
__global__ void gather_kernel(const void* __restrict__ Hv,
                              const float* __restrict__ a_simi,
                              const float* __restrict__ t_simi,
                              const float* __restrict__ adj,
                              unsigned short* __restrict__ X,
                              int S, int D, long long total) {
  long long idx = (long long)blockIdx.x * blockDim.x + threadIdx.x;
  if (idx >= total) return;
  const int C = D >> 2;
  int c = (int)(idx % C); long long r = idx / C;
  int k = (int)(r % S);   r /= S;
  int l = (int)(r % S);
  int b = (int)(r / S);

  const float* Hf = (const float*)Hv;
  const unsigned short* Hb = (const unsigned short*)Hv;

  const long long rowD  = (long long)D;
  const long long rowSD = (long long)S * D;
  const long long base  = (((long long)b * S + l) * S + k) * D + (long long)c * 4;
  const long long dbase = (((long long)b * S + l) * S + l) * D + (long long)c * 4;

  float4 hc = IN_BF16 ? ld4bf(Hb + base)  : ld4f(Hf + base);
  float4 hd = IN_BF16 ? ld4bf(Hb + dbase) : ld4f(Hf + dbase);
  float adjv = adj[((long long)b * S + l) * S + k];

  float4 out;
  out.x = 2.f * hc.x; out.y = 2.f * hc.y; out.z = 2.f * hc.z; out.w = 2.f * hc.w;
  fma4(out, adjv, hd);
  if (k > 0) {
    float4 hm = IN_BF16 ? ld4bf(Hb + base - rowD) : ld4f(Hf + base - rowD);
    fma4(out, t_simi[b * S + k - 1], hm);
  }
  if (k < S - 1) {
    float4 hp = IN_BF16 ? ld4bf(Hb + base + rowD) : ld4f(Hf + base + rowD);
    fma4(out, t_simi[b * S + k + 1], hp);
  }
  if (l > 0) {
    float4 hu = IN_BF16 ? ld4bf(Hb + base - rowSD) : ld4f(Hf + base - rowSD);
    fma4(out, a_simi[b * S + l - 1], hu);
  }
  if (l < S - 1) {
    float4 hdn = IN_BF16 ? ld4bf(Hb + base + rowSD) : ld4f(Hf + base + rowSD);
    fma4(out, a_simi[b * S + l + 1], hdn);
  }

  unsigned long long p = (unsigned long long)f2bf(out.x)
                       | ((unsigned long long)f2bf(out.y) << 16)
                       | ((unsigned long long)f2bf(out.z) << 32)
                       | ((unsigned long long)f2bf(out.w) << 48);
  *(unsigned long long*)(X + base) = p;
}

// ---------- fused GEMM + bias + T5 RMS-norm + ReLU ----------
// Block: 256 threads (8 waves). Each block computes rows [m0, m0+16) x full N.
// Wave w covers n-tiles [w*NTPW, (w+1)*NTPW). K and M are multiples of 32/16.
template <int NTPW, bool OUT_BF16>
__global__ __launch_bounds__(256)
void gemm_rms_relu_kernel(const unsigned short* __restrict__ A,  // bf16 [M,K] row-major
                          const unsigned short* __restrict__ Wp, // packed bf16 B fragments
                          const float* __restrict__ bias,
                          const float* __restrict__ gain,
                          void* __restrict__ Out,
                          int K, int N) {
  __shared__ float ssq[16];

  const int lane = threadIdx.x & 31;
  const int wave = threadIdx.x >> 5;
  const int half = lane >> 4;     // 0: K {0-7,16-23}, rows 0-7 of C ; 1: K {8-15,24-31}, rows 8-15
  const int lid  = lane & 15;     // A: row-in-tile ; B/C/D: column-in-tile
  const int m0   = blockIdx.x * 16;

  v8f acc[NTPW];
  #pragma unroll
  for (int t = 0; t < NTPW; ++t)
    #pragma unroll
    for (int i = 0; i < 8; ++i) acc[t][i] = 0.f;

  const unsigned short* aRow = A + (long long)(m0 + lid) * K + half * 8;
  const int KT = K >> 5;

  for (int kt = 0; kt < KT; ++kt) {
    const unsigned short* ap = aRow + kt * 32;
    v8bf a0 = *(const v8bf*)(ap);
    v8bf a1 = *(const v8bf*)(ap + 16);
    v16bf af;
    #pragma unroll
    for (int i = 0; i < 8; ++i) { af[i] = a0[i]; af[i + 8] = a1[i]; }

    #pragma unroll
    for (int t = 0; t < NTPW; ++t) {
      const int n0 = (wave * NTPW + t) * 16;
      const unsigned short* bp =
          Wp + (((long long)(kt * 2 + half) * N) + n0 + lid) * 16;
      v8bf b0 = *(const v8bf*)(bp);
      v8bf b1 = *(const v8bf*)(bp + 8);
      v16bf bf;
      #pragma unroll
      for (int i = 0; i < 8; ++i) { bf[i] = b0[i]; bf[i + 8] = b1[i]; }
      acc[t] = __builtin_amdgcn_wmma_f32_16x16x32_bf16(
          /*neg_a=*/false, af, /*neg_b=*/false, bf,
          /*c_mod=*/(short)0, acc[t], /*reuse_a=*/false, /*reuse_b=*/false);
    }
  }

  // bias + per-row sum of squares (lane holds 8 rows x NTPW columns)
  float ps[8];
  #pragma unroll
  for (int v = 0; v < 8; ++v) ps[v] = 0.f;
  #pragma unroll
  for (int t = 0; t < NTPW; ++t) {
    const int n = (wave * NTPW + t) * 16 + lid;
    const float bn = bias[n];
    #pragma unroll
    for (int v = 0; v < 8; ++v) {
      float z = acc[t][v] + bn;
      acc[t][v] = z;
      ps[v] = fmaf(z, z, ps[v]);
    }
  }
  // reduce over the 16 lanes of each half (bits 0..3), halves stay independent
  #pragma unroll
  for (int m = 1; m < 16; m <<= 1)
    #pragma unroll
    for (int v = 0; v < 8; ++v) ps[v] += __shfl_xor(ps[v], m, 32);

  if (threadIdx.x < 16) ssq[threadIdx.x] = 0.f;
  __syncthreads();
  if (lid == 0) {
    #pragma unroll
    for (int v = 0; v < 8; ++v) atomicAdd(&ssq[half * 8 + v], ps[v]);
  }
  __syncthreads();

  float rinv[8];
  #pragma unroll
  for (int v = 0; v < 8; ++v)
    rinv[v] = rsqrtf(ssq[half * 8 + v] / (float)N + 1e-12f);

  // gain * norm, ReLU, store
  #pragma unroll
  for (int t = 0; t < NTPW; ++t) {
    const int n = (wave * NTPW + t) * 16 + lid;
    const float g = gain[n];
    #pragma unroll
    for (int v = 0; v < 8; ++v) {
      const long long m = m0 + half * 8 + v;
      float y = fmaxf(acc[t][v] * rinv[v] * g, 0.f);
      if (OUT_BF16) ((unsigned short*)Out)[m * N + n] = f2bf(y);
      else          ((float*)Out)[m * N + n] = y;
    }
  }
}

extern "C" void kernel_launch(void* const* d_in, const int* in_sizes, int n_in,
                              void* d_out, int out_size, void* d_ws, size_t ws_size,
                              hipStream_t stream) {
  (void)in_sizes; (void)n_in; (void)out_size; (void)ws_size;
  const float* table  = (const float*)d_in[0];
  const float* a_simi = (const float*)d_in[1];
  const float* t_simi = (const float*)d_in[2];
  const float* adj    = (const float*)d_in[3];
  const float* W1     = (const float*)d_in[4];
  const float* b1     = (const float*)d_in[5];
  const float* g1     = (const float*)d_in[6];
  const float* W2     = (const float*)d_in[7];
  const float* b2     = (const float*)d_in[8];
  const float* g2     = (const float*)d_in[9];

  const int B = 8, S = 96, Din = 768, Dout = 256;
  const long long M = (long long)B * S * S;  // 73728, multiple of 16

  // workspace carve-out (256B aligned)
  char* ws = (char*)d_ws;
  size_t off = 0;
  auto carve = [&](size_t bytes) -> void* {
    void* p = ws + off;
    off += (bytes + 255) & ~(size_t)255;
    return p;
  };
  unsigned short* X1  = (unsigned short*)carve((size_t)M * Din * 2);
  unsigned short* h1  = (unsigned short*)carve((size_t)M * Din * 2);
  unsigned short* X2  = (unsigned short*)carve((size_t)M * Din * 2);
  unsigned short* Wp1 = (unsigned short*)carve((size_t)Din * Din * 2);
  unsigned short* Wp2 = (unsigned short*)carve((size_t)Din * Dout * 2);

  // 1) repack weights to WMMA B-fragment layout (bf16)
  {
    int tot1 = Din * Din;
    pack_w_kernel<<<(tot1 + 255) / 256, 256, 0, stream>>>(W1, Wp1, Din, Din, tot1);
    int tot2 = Din * Dout;
    pack_w_kernel<<<(tot2 + 255) / 256, 256, 0, stream>>>(W2, Wp2, Din, Dout, tot2);
  }

  const long long gthreads = M * (Din / 4);                 // 14,155,776
  const int gblocks = (int)((gthreads + 255) / 256);        // 55,296

  // 2) X1 = gether(table)+table  (fp32 in -> bf16 out)
  gather_kernel<false><<<gblocks, 256, 0, stream>>>(
      table, a_simi, t_simi, adj, X1, S, Din, gthreads);

  // 3) h1 = relu(rmsnorm(X1 @ W1 + b1) * g1)   N=768 -> 6 n-tiles/wave, bf16 out
  gemm_rms_relu_kernel<6, true><<<(int)(M / 16), 256, 0, stream>>>(
      X1, Wp1, b1, g1, h1, Din, Din);

  // 4) X2 = gether(h1)+h1  (bf16 in -> bf16 out)
  gather_kernel<true><<<gblocks, 256, 0, stream>>>(
      h1, a_simi, t_simi, adj, X2, S, Din, gthreads);

  // 5) out = relu(rmsnorm(X2 @ W2 + b2) * g2)  N=256 -> 2 n-tiles/wave, fp32 out
  gemm_rms_relu_kernel<2, false><<<(int)(M / 16), 256, 0, stream>>>(
      X2, Wp2, b2, g2, d_out, Din, Dout);
}
